// LinearBlock_32006096290592
// MI455X (gfx1250) — matrix operand order repaired and checked by reading, not compiled, not run on hardware
//
#include <hip/hip_runtime.h>

// ---- problem constants (from reference setup_inputs) ----
#define BB 32    // batch
#define II 512   // in features (K)
#define OO 720   // out features (N)
#define FF 321   // timesteps / BN channels (GEMM batch)

#define NCHUNK 128          // N columns per workgroup (8 waves x 16)
#define LDS_PITCH 520       // II + 8 bf16 pad -> 1040B row pitch, bank stride 4

typedef __attribute__((ext_vector_type(16))) __bf16 v16bf;
typedef __attribute__((ext_vector_type(8)))  __bf16 v8bf;
typedef __attribute__((ext_vector_type(8)))  float  v8f;
typedef __attribute__((ext_vector_type(4)))  float  v4f;

__device__ __forceinline__ v16bf cvt16(v4f b0, v4f b1, v4f b2, v4f b3) {
    v16bf r;
    #pragma unroll
    for (int e = 0; e < 4; ++e) {
        r[e]      = (__bf16)b0[e];
        r[4 + e]  = (__bf16)b1[e];
        r[8 + e]  = (__bf16)b2[e];
        r[12 + e] = (__bf16)b3[e];
    }
    return r;
}

// One workgroup: one f-slice, 128 N columns, full M=32.
// 8 wave32s; each wave: one 16-wide N column, two 16-row M tiles.
__global__ __launch_bounds__(256) void bn_bgemm_wmma_kernel(
    const float* __restrict__ x,      // [BB, II, FF]
    const float* __restrict__ W,      // [FF, OO, II]
    const float* __restrict__ bias,   // [FF, OO]
    const float* __restrict__ gamma,  // [FF]
    const float* __restrict__ beta,   // [FF]
    const float* __restrict__ rmean,  // [FF]
    const float* __restrict__ rvar,   // [FF]
    float* __restrict__ out)          // [BB, OO, FF]
{
    __shared__ __bf16 As[BB * LDS_PITCH];   // 33,280 B

    const int f      = blockIdx.x;   // 0..320
    const int nchunk = blockIdx.y;   // 0..5
    const int tid    = threadIdx.x;  // 0..255

    // BatchNorm (eval) per-f scalars: xn = x*inv + shift
    const float inv   = gamma[f] * rsqrtf(rvar[f] + 1e-5f);
    const float shift = beta[f] - rmean[f] * inv;

    // ---- stage A = BN(x[:, :, f]) as bf16 into LDS [32 x 512 (+pad)] ----
    {
        const int row  = tid >> 3;          // 0..31  (b index)
        const int col0 = (tid & 7) * 64;    // 0..448 (i start)
        const float* xp = x + (size_t)row * II * FF + f;
        #pragma unroll
        for (int j = 0; j < 64; j += 8) {
            v8bf pack;
            #pragma unroll
            for (int e = 0; e < 8; ++e) {
                float v = xp[(size_t)(col0 + j + e) * FF] * inv + shift;
                pack[e] = (__bf16)v;
            }
            *(v8bf*)(&As[row * LDS_PITCH + col0 + j]) = pack;   // ds_store_b128
        }
    }
    __syncthreads();

    const int wave = tid >> 5;      // 0..7  (wave32)
    const int lane = tid & 31;
    const int ln   = lane & 15;     // N within tile / M row within half
    const int lh   = lane >> 4;     // half-wave select
    const int o16  = nchunk * NCHUNK + wave * 16;
    if (o16 >= OO) return;          // wave-uniform: EXEC stays all-ones for WMMA

    // B: lane ln loads row (o16+ln) of W_f; lanes<16 read K {0..7, 16..23}, lanes>=16 +8
    const float* wp = W + ((size_t)f * OO + o16 + ln) * II + lh * 8;

    // A: 16-bit A layout -> lanes<16 read K {0..7, 16..23}, lanes>=16 +8
    const __bf16* a0 = &As[(ln)      * LDS_PITCH + lh * 8];
    const __bf16* a1 = &As[(16 + ln) * LDS_PITCH + lh * 8];

    v8f acc0 = {};
    v8f acc1 = {};

    // ---- software-pipelined K loop: raw fp32 B double-buffered so the
    //      bf16 cvts never WAR the B registers of the in-flight WMMA ----
    v4f r0 = *(const v4f*)(wp + 0);
    v4f r1 = *(const v4f*)(wp + 4);
    v4f r2 = *(const v4f*)(wp + 16);
    v4f r3 = *(const v4f*)(wp + 20);

    #pragma unroll 2
    for (int k0 = 0; k0 < II - 32; k0 += 32) {
        // prefetch 4 iterations ahead (RT/near-cache path)
        __builtin_prefetch((const void*)(wp + k0 + 128), 0, 3);

        // issue next K-step's B loads into a fresh register set
        const int kn = k0 + 32;
        v4f n0 = *(const v4f*)(wp + kn);
        v4f n1 = *(const v4f*)(wp + kn + 4);
        v4f n2 = *(const v4f*)(wp + kn + 16);
        v4f n3 = *(const v4f*)(wp + kn + 20);

        // convert current B (registers loaded last iteration)
        v16bf bm = cvt16(r0, r1, r2, r3);

        // A operands from LDS: two b128 loads per M tile
        v8bf alo0 = *(const v8bf*)(a0 + k0);
        v8bf ahi0 = *(const v8bf*)(a0 + k0 + 16);
        v8bf alo1 = *(const v8bf*)(a1 + k0);
        v8bf ahi1 = *(const v8bf*)(a1 + k0 + 16);
        v16bf am0, am1;
        #pragma unroll
        for (int e = 0; e < 8; ++e) {
            am0[e] = alo0[e]; am0[8 + e] = ahi0[e];
            am1[e] = alo1[e]; am1[8 + e] = ahi1[e];
        }

        acc0 = __builtin_amdgcn_wmma_f32_16x16x32_bf16(
                   false, am0, false, bm, (short)0, acc0, false, false);
        acc1 = __builtin_amdgcn_wmma_f32_16x16x32_bf16(
                   false, am1, false, bm, (short)0, acc1, false, false);

        r0 = n0; r1 = n1; r2 = n2; r3 = n3;   // rotate (renamed by unroll)
    }

    // ---- final K step (no next loads) ----
    {
        const int k0 = II - 32;
        v16bf bm = cvt16(r0, r1, r2, r3);
        v8bf alo0 = *(const v8bf*)(a0 + k0);
        v8bf ahi0 = *(const v8bf*)(a0 + k0 + 16);
        v8bf alo1 = *(const v8bf*)(a1 + k0);
        v8bf ahi1 = *(const v8bf*)(a1 + k0 + 16);
        v16bf am0, am1;
        #pragma unroll
        for (int e = 0; e < 8; ++e) {
            am0[e] = alo0[e]; am0[8 + e] = ahi0[e];
            am1[e] = alo1[e]; am1[8 + e] = ahi1[e];
        }
        acc0 = __builtin_amdgcn_wmma_f32_16x16x32_bf16(
                   false, am0, false, bm, (short)0, acc0, false, false);
        acc1 = __builtin_amdgcn_wmma_f32_16x16x32_bf16(
                   false, am1, false, bm, (short)0, acc1, false, false);
    }

    // ---- epilogue: bias (uniform per lane over M) + scatter store ----
    const float bb = bias[(size_t)f * OO + o16 + ln];
    #pragma unroll
    for (int r = 0; r < 8; ++r) {
        const int m = r + lh * 8;   // C layout: lanes<16 M=r, lanes>=16 M=8+r
        out[((size_t)(m)      * OO + o16 + ln) * FF + f] = acc0[r] + bb;
        out[((size_t)(m + 16) * OO + o16 + ln) * FF + f] = acc1[r] + bb;
    }
}

extern "C" void kernel_launch(void* const* d_in, const int* in_sizes, int n_in,
                              void* d_out, int out_size, void* d_ws, size_t ws_size,
                              hipStream_t stream) {
    (void)in_sizes; (void)n_in; (void)d_ws; (void)ws_size; (void)out_size;
    const float* x     = (const float*)d_in[0];
    const float* W     = (const float*)d_in[1];
    const float* bias  = (const float*)d_in[2];
    const float* gamma = (const float*)d_in[3];
    const float* beta  = (const float*)d_in[4];
    const float* rmean = (const float*)d_in[5];
    const float* rvar  = (const float*)d_in[6];
    float* out = (float*)d_out;

    dim3 grid(FF, (OO + NCHUNK - 1) / NCHUNK);  // 321 x 6
    dim3 block(256);
    bn_bgemm_wmma_kernel<<<grid, block, 0, stream>>>(
        x, W, bias, gamma, beta, rmean, rvar, out);
}
